// Hybrid_83451214561395
// MI455X (gfx1250) — compile-verified, matrix operand order checked by reading, not run on hardware
//
#include <hip/hip_runtime.h>
#include <hip/hip_bf16.h>

typedef __attribute__((ext_vector_type(16))) _Float16 v16h;
typedef __attribute__((ext_vector_type(8)))  _Float16 v8h;
typedef __attribute__((ext_vector_type(2)))  _Float16 v2h;
typedef __attribute__((ext_vector_type(8)))  float    v8f;

#define Dm   128
#define Nn   512
#define Bb   2
#define NLay 2
#define Ss   4
#define Vv   20
#define Mrows (Bb * Nn)

#if defined(__has_builtin)
#if __has_builtin(__builtin_amdgcn_global_load_async_to_lds_b32)
#define USE_ASYNC_LDS 1
#endif
#endif

__device__ __forceinline__ float gelu_f(float x) {
    return 0.5f * x * (1.0f + erff(x * 0.70710678118654752440f));
}

// Assemble one 16x16x32 f16 fragment: two contiguous 8-half (16B) LDS runs.
// A-frag: p = &tile[row*ld + k0 + hi*8]   (row-major M x K tile)
// B-frag: p = &tileT[col*ld + k0 + hi*8]  (transposed, col-major N x K tile)
__device__ __forceinline__ v16h frag_ld(const _Float16* p) {
    v8h a = *(const v8h*)p;
    v8h b = *(const v8h*)(p + 16);
    return __builtin_shufflevector(a, b, 0, 1, 2, 3, 4, 5, 6, 7,
                                   8, 9, 10, 11, 12, 13, 14, 15);
}

__device__ __forceinline__ v8f wmma_f16(v16h a, v16h b, v8f c) {
    return __builtin_amdgcn_wmma_f32_16x16x32_f16(
        false, a, false, b, (short)0, c, false, false);
}

// ---------------------------------------------------------------------------
// WMMA GEMM: C[M,N] = act(A[M,K] @ W[K,N] + bias)
// 128 threads = 4 waves; block tile 64(M) x 32(N); wave tile 16x32 (2 WMMA/k).
// ---------------------------------------------------------------------------
#define SA_LD 40   // padded K-stride (halfs), multiple of 8 -> 16B alignment
#define SW_LD 40

__global__ void gemm_wmma_kernel(const float* __restrict__ A,
                                 const float* __restrict__ W,
                                 const float* __restrict__ bias,
                                 float* __restrict__ C,
                                 int M, int N, int K, int act) {
    __shared__ __align__(16) _Float16 sA[64 * SA_LD];   // [64][k 0..31]
    __shared__ __align__(16) _Float16 sWt[32 * SW_LD];  // [n 0..31][k 0..31]
    const int tid  = threadIdx.x;
    const int wave = tid >> 5;
    const int lane = tid & 31;
    const int hi   = lane >> 4;
    const int lo   = lane & 15;
    const int m0 = blockIdx.y * 64;
    const int n0 = blockIdx.x * 32;
    const float* Arow  = A + (size_t)m0 * K;
    const float* Wbase = W + n0;

    v8f acc0 = {}, acc1 = {};
    for (int k0 = 0; k0 < K; k0 += 32) {
        // stage A tile (64x32): 8 independent float2 loads per thread, then
        // packed f16 stores (constant trip count -> full unroll, batched vmem)
        {
            float2 f[8];
#pragma unroll
            for (int it = 0; it < 8; ++it) {
                int p = tid + it * 128;
                int r = p >> 4, c2 = (p & 15) * 2;
                f[it] = *(const float2*)(Arow + r * K + k0 + c2);
            }
#pragma unroll
            for (int it = 0; it < 8; ++it) {
                int p = tid + it * 128;
                int r = p >> 4, c2 = (p & 15) * 2;
                v2h h; h[0] = (_Float16)f[it].x; h[1] = (_Float16)f[it].y;
                *(v2h*)&sA[r * SA_LD + c2] = h;
            }
        }
        // stage W tile transposed: sWt[n][k] = W[k0+k][n0+n]
        {
            float f[8];
#pragma unroll
            for (int it = 0; it < 8; ++it) {
                int p = tid + it * 128;
                int kk = p >> 5, nn = p & 31;
                f[it] = Wbase[(k0 + kk) * N + nn];
            }
#pragma unroll
            for (int it = 0; it < 8; ++it) {
                int p = tid + it * 128;
                int kk = p >> 5, nn = p & 31;
                sWt[nn * SW_LD + kk] = (_Float16)f[it];
            }
        }
        if (k0 + 32 < K)
            __builtin_prefetch(Arow + (tid & 63) * K + k0 + 32, 0, 0);
        __syncthreads();
        v16h af  = frag_ld(&sA[(wave * 16 + lo) * SA_LD + hi * 8]);
        v16h bf0 = frag_ld(&sWt[lo * SW_LD + hi * 8]);
        v16h bf1 = frag_ld(&sWt[(16 + lo) * SW_LD + hi * 8]);
        acc0 = wmma_f16(af, bf0, acc0);
        acc1 = wmma_f16(af, bf1, acc1);
        __syncthreads();
    }
    const int n = n0 + lo;
    const float b0 = bias ? bias[n] : 0.0f;
    const float b1 = bias ? bias[n + 16] : 0.0f;
#pragma unroll
    for (int v = 0; v < 8; ++v) {
        int m = m0 + wave * 16 + hi * 8 + v;   // C/D layout: row = v + 8*hi
        float r0 = acc0[v] + b0;
        float r1 = acc1[v] + b1;
        if (act == 1) { r0 = gelu_f(r0); r1 = gelu_f(r1); }
        float* crow = C + (size_t)m * N;
        crow[n] = r0;
        crow[n + 16] = r1;
    }
}

// ---------------------------------------------------------------------------
// Pair pass 1: per (b,i): w_pre[b,i,j] = ||gelu(a_i+b_j+pb1) @ pw2|| (j<=i),
// -1e9 otherwise.  pw2 cached transposed in LDS; pr stays in registers.
// ---------------------------------------------------------------------------
#define WT_LD 136   // padded k-stride for transposed pw2 (16B-aligned rows)
#define H_LD  136   // padded k-stride for per-wave h tiles

__global__ void pair_norm_kernel(const float* __restrict__ a,
                                 const float* __restrict__ bbuf,
                                 const float* __restrict__ pb1,
                                 const float* __restrict__ pw2,
                                 float* __restrict__ wpre) {
    __shared__ __align__(16) _Float16 sWt[Dm * WT_LD];   // [d'][d]
    __shared__ float    apre[Dm];
    __shared__ __align__(16) _Float16 sH[4][16 * H_LD];
    const int b = blockIdx.y, i = blockIdx.x;
    const int tid  = threadIdx.x;
    const int wave = tid >> 5;
    const int lane = tid & 31;
    const int hi   = lane >> 4;
    const int lo   = lane & 15;

    // stage pw2 transposed (coalesced global reads, batched in groups of 8)
#pragma unroll 8
    for (int it = 0; it < (Dm * Dm) / 128; ++it) {
        int idx = tid + it * 128;
        int d = idx >> 7, dp = idx & 127;
        sWt[dp * WT_LD + d] = (_Float16)pw2[idx];
    }
    if (tid < Dm) apre[tid] = a[((size_t)b * Nn + i) * Dm + tid] + pb1[tid];
    __syncthreads();

    float* wrow = wpre + ((size_t)b * Nn + i) * Nn;
    const float* bbase = bbuf + (size_t)b * Nn * Dm;
    for (int jt = wave; jt < Nn / 16; jt += 4) {
        const int j0 = jt * 16;
        if (j0 > i) {                       // fully masked tile: skip matmul
            if (lo == 0) {
#pragma unroll
                for (int v = 0; v < 8; ++v) wrow[j0 + hi * 8 + v] = -1e9f;
            }
            continue;
        }
        _Float16* H = sH[wave];
        const float* brow = bbase + j0 * Dm;
#pragma unroll 8
        for (int it = 0; it < 32; ++it) {
            int p = lane + it * 32;
            int r = p >> 6, c2 = (p & 63) * 2;
            float2 f = *(const float2*)(brow + r * Dm + c2);
            v2h h;
            h[0] = (_Float16)gelu_f(apre[c2] + f.x);
            h[1] = (_Float16)gelu_f(apre[c2 + 1] + f.y);
            *(v2h*)&H[r * H_LD + c2] = h;
        }
        // A fragments (4 k-steps), each reused across all 8 column tiles
        v16h afr[4];
#pragma unroll
        for (int ks = 0; ks < 4; ++ks)
            afr[ks] = frag_ld(&H[lo * H_LD + ks * 32 + hi * 8]);
        v8f ss = {};
#pragma unroll
        for (int nt = 0; nt < 8; ++nt) {
            const _Float16* wt = &sWt[(nt * 16 + lo) * WT_LD + hi * 8];
            v8f acc = {};
#pragma unroll
            for (int ks = 0; ks < 4; ++ks)
                acc = wmma_f16(afr[ks], frag_ld(wt + ks * 32), acc);
#pragma unroll
            for (int v = 0; v < 8; ++v) ss[v] += acc[v] * acc[v];
        }
        // reduce squares across the 16 lanes of each C half
#pragma unroll
        for (int v = 0; v < 8; ++v) {
            float s = ss[v];
            s += __shfl_xor(s, 1);
            s += __shfl_xor(s, 2);
            s += __shfl_xor(s, 4);
            s += __shfl_xor(s, 8);
            ss[v] = s;
        }
        if (lo == 0) {
#pragma unroll
            for (int v = 0; v < 8; ++v) {
                int j = j0 + hi * 8 + v;
                wrow[j] = (j <= i) ? sqrtf(ss[v] + 1e-12f) : -1e9f;
            }
        }
    }
}

// ---------------------------------------------------------------------------
// In-place row softmax over last dim (Nn). One block (256 thr) per row.
// ---------------------------------------------------------------------------
__global__ void softmax_row_kernel(float* __restrict__ w) {
    float* row = w + (size_t)blockIdx.x * Nn;
    const int tid = threadIdx.x;
    __shared__ float sred[8];
    float m = -3e38f;
#pragma unroll
    for (int it = 0; it < Nn / 256; ++it) m = fmaxf(m, row[tid + it * 256]);
    for (int off = 16; off > 0; off >>= 1) m = fmaxf(m, __shfl_xor(m, off));
    if ((tid & 31) == 0) sred[tid >> 5] = m;
    __syncthreads();
    m = sred[0];
#pragma unroll
    for (int k = 1; k < 8; ++k) m = fmaxf(m, sred[k]);
    __syncthreads();
    float s = 0.0f;
#pragma unroll
    for (int it = 0; it < Nn / 256; ++it) s += expf(row[tid + it * 256] - m);
    for (int off = 16; off > 0; off >>= 1) s += __shfl_xor(s, off);
    if ((tid & 31) == 0) sred[tid >> 5] = s;
    __syncthreads();
    float tot = 0.0f;
#pragma unroll
    for (int k = 0; k < 8; ++k) tot += sred[k];
    const float inv = 1.0f / tot;
#pragma unroll
    for (int it = 0; it < Nn / 256; ++it) {
        int j = tid + it * 256;
        row[j] = expf(row[j] - m) * inv;
    }
}

// ---------------------------------------------------------------------------
// Pair pass 2: hbar[b,i,:] = sum_{j<=i} w[b,i,j] * gelu(a_i + b_j + pb1)
// Weight row staged through the CDNA5 async global->LDS path when available.
// ---------------------------------------------------------------------------
__global__ void pair_wsum_kernel(const float* __restrict__ a,
                                 const float* __restrict__ bbuf,
                                 const float* __restrict__ pb1,
                                 const float* __restrict__ w,
                                 float* __restrict__ hbar) {
    __shared__ float sw[Nn];
    const int b = blockIdx.y, i = blockIdx.x, d = threadIdx.x;
    const float* wsrc = w + ((size_t)b * Nn + i) * Nn;
#ifdef USE_ASYNC_LDS
#pragma unroll
    for (int it = 0; it < Nn / 128; ++it) {
        int j = threadIdx.x + it * 128;
        __builtin_amdgcn_global_load_async_to_lds_b32(
            (__attribute__((address_space(1))) int*)(wsrc + j),
            (__attribute__((address_space(3))) int*)(&sw[j]), 0, 0);
    }
    asm volatile("s_wait_asynccnt 0x0" ::: "memory");
#else
#pragma unroll
    for (int it = 0; it < Nn / 128; ++it) {
        int j = threadIdx.x + it * 128;
        sw[j] = wsrc[j];
    }
#endif
    __syncthreads();
    const float apre = a[((size_t)b * Nn + i) * Dm + d] + pb1[d];
    const float* bcol = bbuf + (size_t)b * Nn * Dm + d;
    float acc = 0.0f;
    for (int j = 0; j <= i; ++j)   // weights are exactly 0 beyond i
        acc += sw[j] * gelu_f(apre + bcol[j * Dm]);
    hbar[((size_t)b * Nn + i) * Dm + d] = acc;
}

// ---------------------------------------------------------------------------
// Triple: hmean[row,:] = mean_s gelu(ta + xb[ji] + xc[ki] + tb1)
// ---------------------------------------------------------------------------
__global__ void triple_gather_kernel(const float* __restrict__ ta,
                                     const float* __restrict__ xb,
                                     const float* __restrict__ xc,
                                     const float* __restrict__ tb1,
                                     const int* __restrict__ ji,
                                     const int* __restrict__ ki,
                                     float* __restrict__ hmean) {
    const int row = blockIdx.x;          // b*Nn + n
    const int b = row / Nn;
    const int d = threadIdx.x;
    const float base = ta[(size_t)row * Dm + d] + tb1[d];
    const float* xbb = xb + (size_t)b * Nn * Dm + d;
    const float* xcb = xc + (size_t)b * Nn * Dm + d;
    const int* jp = ji + (size_t)row * Ss;
    const int* kp = ki + (size_t)row * Ss;
    float acc = 0.0f;
#pragma unroll
    for (int s = 0; s < Ss; ++s)
        acc += gelu_f(base + xbb[jp[s] * Dm] + xcb[kp[s] * Dm]);
    hmean[(size_t)row * Dm + d] = acc * (1.0f / Ss);
}

// ---------------------------------------------------------------------------
__global__ void gate_kernel(const float* __restrict__ x,
                            const float* __restrict__ pp,
                            const float* __restrict__ tt,
                            const float* __restrict__ gp,
                            const float* __restrict__ gt,
                            const float* __restrict__ gb,
                            float* __restrict__ u) {
    const size_t idx = (size_t)blockIdx.x * Dm + threadIdx.x;
    const float gv = gp[idx] + gt[idx] + gb[threadIdx.x];
    const float s = 1.0f / (1.0f + expf(-gv));
    u[idx] = x[idx] + s * pp[idx] + (1.0f - s) * tt[idx];
}

__global__ void ln_kernel(const float* __restrict__ xin,
                          const float* __restrict__ res,
                          const float* __restrict__ g,
                          const float* __restrict__ beta,
                          float* __restrict__ out) {
    const int row = blockIdx.x, tid = threadIdx.x;
    __shared__ float sred[4];
    float v = xin[(size_t)row * Dm + tid];
    if (res) v += res[(size_t)row * Dm + tid];
    float s = v;
    for (int off = 16; off > 0; off >>= 1) s += __shfl_xor(s, off);
    if ((tid & 31) == 0) sred[tid >> 5] = s;
    __syncthreads();
    const float mean = (sred[0] + sred[1] + sred[2] + sred[3]) * (1.0f / Dm);
    __syncthreads();
    const float dv = v - mean;
    float s2 = dv * dv;
    for (int off = 16; off > 0; off >>= 1) s2 += __shfl_xor(s2, off);
    if ((tid & 31) == 0) sred[tid >> 5] = s2;
    __syncthreads();
    const float var = (sred[0] + sred[1] + sred[2] + sred[3]) * (1.0f / Dm);
    out[(size_t)row * Dm + tid] = dv * rsqrtf(var + 1e-5f) * g[tid] + beta[tid];
}

__global__ void embed_kernel(const int* __restrict__ ids,
                             const float* __restrict__ emb,
                             const float* __restrict__ pos,
                             float* __restrict__ x) {
    const int row = blockIdx.x, d = threadIdx.x;
    const int n = row % Nn;
    x[(size_t)row * Dm + d] = emb[(size_t)ids[row] * Dm + d] + pos[(size_t)n * Dm + d];
}

__global__ void head_kernel(const float* __restrict__ x,
                            const float* __restrict__ hw,
                            const float* __restrict__ hb,
                            float* __restrict__ out) {
    const int row = blockIdx.x, v = threadIdx.x;
    if (v >= Vv) return;
    const float* xr = x + (size_t)row * Dm;
    float acc = hb[v];
    for (int d = 0; d < Dm; ++d) acc += xr[d] * hw[d * Vv + v];
    out[(size_t)row * Vv + v] = acc;
}

// ---------------------------------------------------------------------------
static inline void gemm(const float* A, const float* W, const float* bias,
                        float* C, int M, int N, int K, int act, hipStream_t s) {
    gemm_wmma_kernel<<<dim3(N / 32, M / 64), 128, 0, s>>>(A, W, bias, C, M, N, K, act);
}

extern "C" void kernel_launch(void* const* d_in, const int* in_sizes, int n_in,
                              void* d_out, int out_size, void* d_ws, size_t ws_size,
                              hipStream_t stream) {
    (void)in_sizes; (void)n_in; (void)out_size; (void)ws_size;
    const int*   ids   = (const int*)d_in[0];
    const int*   ji_a  = (const int*)d_in[1];
    const int*   ki_a  = (const int*)d_in[2];
    const float* emb   = (const float*)d_in[3];
    const float* pos   = (const float*)d_in[4];
    const float* hw    = (const float*)d_in[5 + 22 * NLay + 0];
    const float* hb    = (const float*)d_in[5 + 22 * NLay + 1];

    float* ws = (float*)d_ws;
    size_t off = 0;
    auto alloc = [&](size_t n) { float* p = ws + off; off += n; return p; };
    const size_t RD = (size_t)Mrows * Dm;
    float* X    = alloc(RD);
    float* A1   = alloc(RD);
    float* B1   = alloc(RD);
    float* HBAR = alloc(RD);
    float* PP   = alloc(RD);
    float* TA   = alloc(RD);
    float* XB   = alloc(RD);
    float* XC   = alloc(RD);
    float* HM   = alloc(RD);
    float* TT   = alloc(RD);
    float* GP   = alloc(RD);
    float* GT   = alloc(RD);
    float* U    = alloc(RD);
    float* F1   = alloc((size_t)Mrows * 4 * Dm);
    float* F2   = alloc(RD);
    float* WN   = alloc((size_t)Bb * Nn * Nn);

    embed_kernel<<<Mrows, Dm, 0, stream>>>(ids, emb, pos, X);

    for (int l = 0; l < NLay; ++l) {
        const int base = 5 + 22 * l;
        #define PRM(k) ((const float*)d_in[base + (k)])
        const float *pw1a = PRM(0),  *pw1b = PRM(1),  *pb1 = PRM(2);
        const float *pw2  = PRM(3),  *pb2  = PRM(4);
        const float *tw1a = PRM(5),  *tw1b = PRM(6),  *tw1c = PRM(7), *tb1 = PRM(8);
        const float *tw2  = PRM(9),  *tb2  = PRM(10);
        const float *gwp  = PRM(11), *gwt  = PRM(12), *gb = PRM(13);
        const float *n1g  = PRM(14), *n1b  = PRM(15);
        const float *fw1  = PRM(16), *fb1  = PRM(17);
        const float *fw2  = PRM(18), *fb2  = PRM(19);
        const float *n2g  = PRM(20), *n2b  = PRM(21);
        #undef PRM

        // ---- pair branch ----
        gemm(X, pw1a, nullptr, A1, Mrows, Dm, Dm, 0, stream);
        gemm(X, pw1b, nullptr, B1, Mrows, Dm, Dm, 0, stream);
        pair_norm_kernel<<<dim3(Nn, Bb), 128, 0, stream>>>(A1, B1, pb1, pw2, WN);
        softmax_row_kernel<<<Bb * Nn, 256, 0, stream>>>(WN);
        pair_wsum_kernel<<<dim3(Nn, Bb), Dm, 0, stream>>>(A1, B1, pb1, WN, HBAR);
        gemm(HBAR, pw2, pb2, PP, Mrows, Dm, Dm, 0, stream);

        // ---- triple branch ----
        gemm(X, tw1a, nullptr, TA, Mrows, Dm, Dm, 0, stream);
        gemm(X, tw1b, nullptr, XB, Mrows, Dm, Dm, 0, stream);
        gemm(X, tw1c, nullptr, XC, Mrows, Dm, Dm, 0, stream);
        triple_gather_kernel<<<Mrows, Dm, 0, stream>>>(
            TA, XB, XC, tb1,
            ji_a + (size_t)l * Bb * Nn * Ss, ki_a + (size_t)l * Bb * Nn * Ss, HM);
        gemm(HM, tw2, tb2, TT, Mrows, Dm, Dm, 0, stream);

        // ---- gate + LN1 ----
        gemm(PP, gwp, nullptr, GP, Mrows, Dm, Dm, 0, stream);
        gemm(TT, gwt, nullptr, GT, Mrows, Dm, Dm, 0, stream);
        gate_kernel<<<Mrows, Dm, 0, stream>>>(X, PP, TT, GP, GT, gb, U);
        ln_kernel<<<Mrows, Dm, 0, stream>>>(U, nullptr, n1g, n1b, X);

        // ---- FFN + LN2 ----
        gemm(X, fw1, fb1, F1, Mrows, 4 * Dm, Dm, 1, stream);
        gemm(F1, fw2, fb2, F2, Mrows, Dm, 4 * Dm, 0, stream);
        ln_kernel<<<Mrows, Dm, 0, stream>>>(X, F2, n2g, n2b, X);
    }

    head_kernel<<<Mrows, 32, 0, stream>>>(X, hw, hb, (float*)d_out);
}
